// RecurrentLayer_18253611008720
// MI455X (gfx1250) — matrix operand order repaired and checked by reading, not compiled, not run on hardware
//
#include <hip/hip_runtime.h>
#include <hip/hip_bf16.h>
#include <math.h>

#define SEQ    64
#define VOCAB  50257
#define LATENT 1024
#define KDIM   (LATENT + SEQ)   // 1088
#define KTILES (KDIM / 32)      // 34 K-tiles of 32
#define MTILES (SEQ / 16)       // 4 M-tiles of 16
#define INV_TEMP 10.0f          // 1 / 0.1
#define NBLK   64               // persistent-kernel blocks (one vocab slice each)

#if __has_builtin(__builtin_amdgcn_tensor_load_to_lds) && \
    __has_builtin(__builtin_amdgcn_s_wait_tensorcnt)
#define HAVE_TDM 1
#else
#define HAVE_TDM 0
#endif

typedef __bf16 v16bf __attribute__((ext_vector_type(16)));
typedef float  v8f   __attribute__((ext_vector_type(8)));
typedef unsigned int u32x4 __attribute__((ext_vector_type(4)));
typedef int          i32x4 __attribute__((ext_vector_type(4)));
typedef int          i32x8 __attribute__((ext_vector_type(8)));

__device__ __forceinline__ unsigned short f32_to_bf16_bits(float f) {
  union { float f; unsigned u; } a; a.f = f;
  unsigned u = a.u;
  u += 0x7fffu + ((u >> 16) & 1u);   // round-to-nearest-even
  return (unsigned short)(u >> 16);
}

__device__ __forceinline__ unsigned mix32(unsigned x) {
  x ^= x >> 16; x *= 0x7feb352dU;
  x ^= x >> 15; x *= 0x846ca68bU;
  x ^= x >> 16; return x;
}

__device__ __forceinline__ float gumbel_hash(unsigned seed, unsigned idx, unsigned v) {
  unsigned h = mix32(seed ^ mix32(idx * 0x9e3779b9U + mix32(v + 0x85ebca6bU)));
  float u = ((float)h + 0.5f) * 2.3283064365386963e-10f;  // (0,1)
  return -__logf(-__logf(u));
}

// ---------------------------------------------------------------------------
// Kernel 1: build A (64 x 1088) latent-window matrix pre-swizzled into the
// CDNA5 16-bit A-fragment layout; also zero the grid-barrier flags.
// ---------------------------------------------------------------------------
__global__ void build_afrag(const float* __restrict__ latent,
                            unsigned short* __restrict__ afrag,
                            unsigned* __restrict__ bar) {
  const int total = MTILES * KTILES * 32 * 16;
  int t = blockIdx.x * blockDim.x + threadIdx.x;
  if (t == 0) { bar[0] = 0u; bar[1] = 0u; }
  if (t >= total) return;
  int i     = t & 15;
  int rest  = t >> 4;
  int l     = rest & 31;
  int rest2 = rest >> 5;
  int kt    = rest2 % KTILES;
  int mt    = rest2 / KTILES;
  int m     = mt * 16 + (l & 15);
  int half  = l >> 4;
  int kl    = (i < 8) ? (half * 8 + i) : (16 + half * 8 + (i - 8));
  int k     = kt * 32 + kl;
  float val = (k >= m && k < m + LATENT) ? latent[k - m] : 0.0f;
  afrag[t] = f32_to_bf16_bits(val);
}

#if HAVE_TDM
// Issue a TDM load of a 2D f32 tile (tile_dim0=16 cols x tile_dim1=32 rows)
// into LDS with 4-dword padding every 16 dwords (row stride = 20 dwords).
// D# packing per CDNA5 ISA ch.8: group0 {count,lds_addr,global_addr,type=2},
// group1 {data_size=4B, pad_enable, pad_interval=16dw, pad_amount=4dw,
// tensor_dim0/1, tile_dim0/1, tensor_dim0_stride}.  tensor_dim0 = remaining
// columns -> hardware zero-fills the ragged tail (no branch needed).
__device__ __forceinline__ void tdm_issue(const float* gptr, unsigned ldsAddr,
                                          unsigned remCols, unsigned remRows) {
  unsigned long long ga = (unsigned long long)(const void*)gptr;
  u32x4 g0 = { 1u,                                   // count=1 valid descriptor
               ldsAddr,                               // lds_addr (bytes)
               (unsigned)ga,                          // global_addr[31:0]
               ((unsigned)(ga >> 32) & 0x01ffffffu) | (2u << 30) };  // addr[56:32]|type=2
  i32x8 g1 = {
    (int)((2u << 16) | (1u << 20) | (3u << 22) | (3u << 25)), // 4B elems, pad 4dw/16dw
    (int)((remCols & 0xffffu) << 16),                         // tensor_dim0 lo16
    (int)(((remCols >> 16) & 0xffffu) | ((remRows & 0xffffu) << 16)),
    (int)(((remRows >> 16) & 0xffffu) | (16u << 16)),         // tile_dim0 = 16
    (int)(32u),                                               // tile_dim1 = 32
    (int)(VOCAB),                                             // dim0_stride lo32
    0, 0 };
  i32x4 z4 = {0, 0, 0, 0};
#if __clang_major__ >= 23
  i32x8 z8 = {0, 0, 0, 0, 0, 0, 0, 0};
  __builtin_amdgcn_tensor_load_to_lds(g0, g1, z4, z4, z8, 0);
#else
  __builtin_amdgcn_tensor_load_to_lds(g0, g1, z4, z4, 0);
#endif
}
#endif

// ---------------------------------------------------------------------------
// Kernel 2: Lat = A @ W (64 x 50257), ONE pass over W (218 MB ~ 10us HBM
// floor).  One wave per 16-column strip; per K-tile, the wave's 32x16 f32 W
// tile is staged LDS-side by the Tensor Data Mover (double-buffered, per-wave
// TENSORcnt -> zero barriers), then fed to v_wmma_f32_16x16x32_bf16.
// ---------------------------------------------------------------------------
__global__ void __launch_bounds__(256)
lat_gemm(const float* __restrict__ Wg, const unsigned short* __restrict__ afrag,
         float* __restrict__ Lat) {
#if HAVE_TDM
  __shared__ float tile[8][2][640];     // 8 waves x 2 bufs x (32 rows * 20 dw)
#endif
  const int lane  = threadIdx.x & 31;
  const int wave  = threadIdx.x >> 5;
  const int strip = blockIdx.x * 8 + wave;
  const int n0    = strip * 16;
  if (n0 >= VOCAB) return;              // wave-uniform exit

  const v8f vzero = {0.f,0.f,0.f,0.f,0.f,0.f,0.f,0.f};
  v8f c[MTILES];
#pragma unroll
  for (int mt = 0; mt < MTILES; ++mt) c[mt] = vzero;

#if HAVE_TDM
  const unsigned ldsBase[2] = {
    (unsigned)(size_t)(void*)&tile[wave][0][0],   // flat addr low 32 = LDS addr
    (unsigned)(size_t)(void*)&tile[wave][1][0] };
  const unsigned remCols = (unsigned)(VOCAB - n0);
  tdm_issue(Wg + (size_t)0 * VOCAB + n0, ldsBase[0], remCols, (unsigned)KDIM);
#endif

  for (int kt = 0; kt < KTILES; ++kt) {
    union { v16bf v; unsigned short s[16]; } bf;
#if HAVE_TDM
    if (kt + 1 < KTILES) {
      tdm_issue(Wg + (size_t)(kt + 1) * 32 * VOCAB + n0, ldsBase[(kt + 1) & 1],
                remCols, (unsigned)(KDIM - (kt + 1) * 32));
      __builtin_amdgcn_s_wait_tensorcnt(1);   // tile kt landed
    } else {
      __builtin_amdgcn_s_wait_tensorcnt(0);
    }
    const float* colp = &tile[wave][kt & 1][lane & 15];  // column n = lane&15, k stride 20 dw
#pragma unroll
    for (int i = 0; i < 16; ++i) {
      const int kk = (i < 8) ? ((lane >> 4) * 8 + i) : (16 + (lane >> 4) * 8 + (i - 8));
      bf.s[i] = f32_to_bf16_bits(colp[kk * 20]);
    }
#else
    int n = n0 + (lane & 15); n = (n < VOCAB) ? n : (VOCAB - 1);  // clamp ragged tail
    const float* wcol = Wg + (size_t)(kt * 32) * VOCAB + n;
#pragma unroll
    for (int i = 0; i < 16; ++i) {
      const int kk = (i < 8) ? ((lane >> 4) * 8 + i) : (16 + (lane >> 4) * 8 + (i - 8));
      bf.s[i] = f32_to_bf16_bits(wcol[(size_t)kk * VOCAB]);
    }
#endif
    const int abase = (kt * 32 + lane) * 16;
#pragma unroll
    for (int mt = 0; mt < MTILES; ++mt) {
      const v16bf a = *(const v16bf*)(afrag + mt * (KTILES * 512) + abase);
      c[mt] = __builtin_amdgcn_wmma_f32_16x16x32_bf16(
          false, a, false, bf.v, (short)0, c[mt], false, false);
    }
  }

  // C/D layout: VGPR r, lane l -> M = r + 8*(l>>4), N = l&15
  const int col = n0 + (lane & 15);
  if (col < VOCAB) {
#pragma unroll
    for (int mt = 0; mt < MTILES; ++mt) {
#pragma unroll
      for (int r = 0; r < 8; ++r) {
        const int row = mt * 16 + r + 8 * (lane >> 4);
        Lat[(size_t)row * VOCAB + col] = c[mt][r];
      }
    }
  }
}

// ---------------------------------------------------------------------------
// Persistent sequential kernel: all 64 autoregressive steps in ONE launch.
// Acc[v] is block-local, so only the 64-entry partials need cross-block sync:
// 2 grid barriers per step (agent-scope atomics + s_sleep spin).
// ---------------------------------------------------------------------------
__device__ __forceinline__ void grid_barrier(unsigned* cnt, unsigned* gen) {
  __threadfence();
  __syncthreads();
  if (threadIdx.x == 0) {
    unsigned g = __hip_atomic_load(gen, __ATOMIC_ACQUIRE, __HIP_MEMORY_SCOPE_AGENT);
    unsigned arrived = __hip_atomic_fetch_add(cnt, 1u, __ATOMIC_ACQ_REL,
                                              __HIP_MEMORY_SCOPE_AGENT) + 1u;
    if (arrived == (unsigned)NBLK) {
      __hip_atomic_store(cnt, 0u, __ATOMIC_RELAXED, __HIP_MEMORY_SCOPE_AGENT);
      __hip_atomic_fetch_add(gen, 1u, __ATOMIC_ACQ_REL, __HIP_MEMORY_SCOPE_AGENT);
    } else {
      while (__hip_atomic_load(gen, __ATOMIC_ACQUIRE, __HIP_MEMORY_SCOPE_AGENT) == g)
        __builtin_amdgcn_s_sleep(1);
    }
  }
  __syncthreads();
}

__device__ __forceinline__ void block_reduce(float* sm, float* ss, float* sy,
                                             int* si, int tid) {
  for (int off = 128; off > 0; off >>= 1) {
    if (tid < off) {
      const float m2 = sm[tid + off], s2 = ss[tid + off];
      const float M  = fmaxf(sm[tid], m2);
      ss[tid] = ss[tid] * __expf(sm[tid] - M) + s2 * __expf(m2 - M);
      sm[tid] = M;
      if (sy[tid + off] > sy[tid]) { sy[tid] = sy[tid + off]; si[tid] = si[tid + off]; }
    }
    __syncthreads();
  }
}

__global__ void __launch_bounds__(256)
seq_sample(const float* __restrict__ Lat, const float* __restrict__ bias,
           const float* __restrict__ Wg, float* __restrict__ Acc,
           float* __restrict__ part, unsigned* __restrict__ bar,
           float* __restrict__ out) {
  __shared__ float sm[256], ss[256], sy[256];
  __shared__ int   si[256];
  __shared__ float bc_lse, bc_tok;
  const int tid   = threadIdx.x;
  const int b     = blockIdx.x;
  const int chunk = (VOCAB + NBLK - 1) / NBLK;           // 786
  const int v0    = b * chunk;
  const int v1    = (v0 + chunk < VOCAB) ? (v0 + chunk) : VOCAB;
  unsigned* cnt = bar;
  unsigned* gen = bar + 1;

  for (int v = v0 + tid; v < v1; v += 256) Acc[v] = 0.0f;  // block-local slice
  __syncthreads();

  for (int idx = 0; idx < SEQ; ++idx) {
    const float* latRow = Lat + (size_t)idx * VOCAB;

    // phase 1: streaming logsumexp + Gumbel-argmax over my slice
    float m = -3.402823466e38f, s = 0.0f, ym = -3.402823466e38f;
    int yi = 0;
    for (int v = v0 + tid; v < v1; v += 256) {
      const float logit = Acc[v] + latRow[v] + bias[v];
      if (logit > m) { s = s * __expf(m - logit) + 1.0f; m = logit; }
      else           { s += __expf(logit - m); }
      const float y = (logit + gumbel_hash(0x12345687u, idx, v)) * INV_TEMP
                      + gumbel_hash(0x89abcdefu, idx, v);
      if (y > ym) { ym = y; yi = v; }
    }
    sm[tid] = m; ss[tid] = s; sy[tid] = ym; si[tid] = yi;
    __syncthreads();
    block_reduce(sm, ss, sy, si, tid);
    if (tid == 0) {
      part[b * 4 + 0] = sm[0];
      part[b * 4 + 1] = ss[0];
      part[b * 4 + 2] = sy[0];
      ((int*)part)[b * 4 + 3] = si[0];
    }
    grid_barrier(cnt, gen);   // publish partials

    // phase 2: every block redundantly reduces all NBLK partials
    if (tid < NBLK) {
      sm[tid] = part[tid * 4 + 0];
      ss[tid] = part[tid * 4 + 1];
      sy[tid] = part[tid * 4 + 2];
      si[tid] = ((const int*)part)[tid * 4 + 3];
    } else {
      sm[tid] = -3.402823466e38f; ss[tid] = 0.0f;
      sy[tid] = -3.402823466e38f; si[tid] = 0;
    }
    __syncthreads();
    block_reduce(sm, ss, sy, si, tid);
    if (tid == 0) {
      bc_lse = sm[0] + __logf(ss[0]);
      bc_tok = (float)si[0];              // sampled token index as float
    }
    __syncthreads();
    const float lse = bc_lse;
    const float tok = bc_tok;
    grid_barrier(cnt, gen);   // partials free for reuse next step

    // phase 3: probs_out[idx] = softmax(logits); Acc += tok * W[idx,:]
    const float* wrow = Wg + (size_t)idx * VOCAB;
    for (int v = v0 + tid; v < v1; v += 256) {
      const float acc   = Acc[v];
      const float logit = acc + latRow[v] + bias[v];
      out[(size_t)idx * VOCAB + v] = __expf(logit - lse);
      Acc[v] = acc + tok * wrow[v];
    }
  }
}

extern "C" void kernel_launch(void* const* d_in, const int* in_sizes, int n_in,
                              void* d_out, int out_size, void* d_ws, size_t ws_size,
                              hipStream_t stream) {
  (void)in_sizes; (void)n_in; (void)out_size; (void)ws_size;
  const float* latent = (const float*)d_in[0];   // (1, 1024) f32
  const float* Wg     = (const float*)d_in[1];   // (1088, 50257) f32
  const float* bias   = (const float*)d_in[2];   // (50257,) f32
  float* out = (float*)d_out;                    // (64, 1, 50257) f32

  char* ws = (char*)d_ws;
  unsigned short* afrag = (unsigned short*)ws;                 // 139,264 B
  size_t off = (size_t)MTILES * KTILES * 32 * 16 * sizeof(unsigned short);
  off = (off + 255) & ~(size_t)255;
  float* Lat = (float*)(ws + off);                             // 12.87 MB
  off += (size_t)SEQ * VOCAB * sizeof(float);
  off = (off + 255) & ~(size_t)255;
  float* Acc = (float*)(ws + off);                             // 201 KB
  off += (size_t)VOCAB * sizeof(float);
  off = (off + 255) & ~(size_t)255;
  float* part = (float*)(ws + off);                            // 1 KB
  off += (size_t)NBLK * 4 * sizeof(float);
  off = (off + 255) & ~(size_t)255;
  unsigned* bar = (unsigned*)(ws + off);                       // 2 words

  const int atotal = MTILES * KTILES * 32 * 16;
  build_afrag<<<(atotal + 255) / 256, 256, 0, stream>>>(latent, afrag, bar);

  const int strips = (VOCAB + 15) / 16;                        // 3142
  lat_gemm<<<(strips + 7) / 8, 256, 0, stream>>>(Wg, afrag, Lat);

  seq_sample<<<NBLK, 256, 0, stream>>>(Lat, bias, Wg, Acc, part, bar, out);
}
